// MyLSTM_42125039239311
// MI455X (gfx1250) — matrix operand order repaired and checked by reading, not compiled, not run on hardware
//
#include <hip/hip_runtime.h>
#include <cstddef>

// ---------------- CDNA5 (gfx1250) vector types ----------------
typedef __attribute__((ext_vector_type(16))) __bf16 v16bf;
typedef __attribute__((ext_vector_type(8)))  float  v8f;
typedef __attribute__((ext_vector_type(2)))  float  v2f;
typedef __attribute__((ext_vector_type(4)))  float  f4;

#define LSTM_BATCH 4096
#define LSTM_SEQ   512
#define LSTM_IN    64
#define LSTM_HID   10

// ---------------- fast transcendentals ----------------
__device__ __forceinline__ float fast_rcp(float x) {
  return __builtin_amdgcn_rcpf(x);
}

#if __has_builtin(__builtin_amdgcn_tanhf)
__device__ __forceinline__ float fast_tanh(float x) {
  return __builtin_amdgcn_tanhf(x);   // v_tanh_f32 (CDNA5 TRANS op)
}
#else
__device__ __forceinline__ float fast_tanh(float x) {
  // tanh(x) = 1 - 2/(exp(2x)+1)
  float e = __expf(2.0f * x);
  return 1.0f - 2.0f * fast_rcp(e + 1.0f);
}
#endif

__device__ __forceinline__ float fast_sigmoid(float x) {
  // sigmoid(x) = 0.5*tanh(x/2) + 0.5  -> single TRANS op per gate
  return __builtin_fmaf(0.5f, fast_tanh(0.5f * x), 0.5f);
}

// =====================================================================
// Fused LSTM:  one wave == one 16-row batch tile, loops over all 512 t.
//   gates[16 x 4*16] = bf16WMMA(x_t[16x64], Wih^T)  (4 gate tiles, K=64)
//                    + f32WMMA(h[16x12], Whh^T)     (K=10 padded to 12)
//                    + bias   (folded into C init)
// Gate g lives in its own 16-col N-tile (cols 0..9 valid) so i/f/g/o of a
// hidden unit land in the same lane & accumulator slot -> pure VALU cell.
// =====================================================================
__global__ __launch_bounds__(32)
void lstm_fused_wmma(const float* __restrict__ x,
                     const float* __restrict__ Wih,
                     const float* __restrict__ Whh,
                     const float* __restrict__ bih,
                     const float* __restrict__ bhh,
                     float* __restrict__ out)
{
  const int lane = threadIdx.x & 31;
  const int h    = lane >> 4;    // lane half (WMMA layout)
  const int n    = lane & 15;    // N column within tile / A row
  const int r    = n;            // batch row within tile (A-matrix role)
  const int b0   = blockIdx.x * 16;

  // ---------------- one-time weight staging into B-layout registers ----
  // Bih[g][ks]: B 32x16 bf16 tile. element e of lane (h,n) = K = ks*32+8h+e+(e&8),
  // value = Wih[g*10+n][K]  (zero-padded for n>=10).
  v16bf Bih[4][2];
  #pragma unroll
  for (int g = 0; g < 4; ++g) {
    #pragma unroll
    for (int ks = 0; ks < 2; ++ks) {
      const float* wp = Wih + (g * 10 + n) * LSTM_IN + ks * 32 + h * 8;
      v16bf bb;
      #pragma unroll
      for (int e = 0; e < 16; ++e)
        bb[e] = (__bf16)((n < 10) ? wp[e + (e & 8)] : 0.0f);
      Bih[g][ks] = bb;
    }
  }

  // Bhh[g][ks]: B 4x16 f32 tile for v_wmma_f32_16x16x4_f32.
  // lane (h,n): v0 = Whh[g*10+n][4ks+2h], v1 = Whh[g*10+n][4ks+2h+1]
  v2f Bhh[4][3];
  #pragma unroll
  for (int g = 0; g < 4; ++g) {
    #pragma unroll
    for (int ks = 0; ks < 3; ++ks) {
      int u0 = 4 * ks + 2 * h;
      v2f bb;
      bb[0] = (n < 10 && u0     < LSTM_HID) ? Whh[(g * 10 + n) * LSTM_HID + u0]     : 0.0f;
      bb[1] = (n < 10 && u0 + 1 < LSTM_HID) ? Whh[(g * 10 + n) * LSTM_HID + u0 + 1] : 0.0f;
      Bhh[g][ks] = bb;
    }
  }

#if !__has_builtin(__builtin_amdgcn_wmma_f32_16x16x4_f32)
  // VALU fallback for the recurrent matmul: per-lane copy of Whh row.
  float WhhR[4][LSTM_HID];
  #pragma unroll
  for (int g = 0; g < 4; ++g)
    #pragma unroll
    for (int u = 0; u < LSTM_HID; ++u)
      WhhR[g][u] = (n < 10) ? Whh[(g * 10 + n) * LSTM_HID + u] : 0.0f;
#endif

  float bias[4];
  #pragma unroll
  for (int g = 0; g < 4; ++g)
    bias[g] = (n < 10) ? (bih[g * 10 + n] + bhh[g * 10 + n]) : 0.0f;

  // ---------------- state ----------------
  float cst[8];                    // cell state, C-layout (rows 8h+j, col n)
  #pragma unroll
  for (int j = 0; j < 8; ++j) cst[j] = 0.0f;

  v2f Ah[3];                       // h in A-layout (16x4 f32 tiles), h0 = 0
  #pragma unroll
  for (int ks = 0; ks < 3; ++ks) { Ah[ks][0] = 0.0f; Ah[ks][1] = 0.0f; }

  __shared__ float hbuf[16][LSTM_HID];   // wave-private h transpose buffer
  for (int idx = lane; idx < 16 * LSTM_HID; idx += 32)
    (&hbuf[0][0])[idx] = 0.0f;
  __syncthreads();

  // x addressing: lane (h,r) reads row b0+r; per-t chunk base
  //   = t*64 + ks*32 + c*16 + h*8   (two f4 loads per chunk, 32B aligned)
  const float* xrow = x + ((size_t)(b0 + r) * LSTM_SEQ) * LSTM_IN;
  const size_t HS_OFF = (size_t)LSTM_BATCH * LSTM_SEQ * LSTM_HID;

  f4 raw[8];                       // double-buffered raw x (idx = ks*4+c*2+p)
  #pragma unroll
  for (int ks = 0; ks < 2; ++ks)
    #pragma unroll
    for (int c = 0; c < 2; ++c) {
      const f4* q = (const f4*)(xrow + ks * 32 + c * 16 + h * 8);
      raw[ks * 4 + c * 2 + 0] = q[0];
      raw[ks * 4 + c * 2 + 1] = q[1];
    }

  for (int t = 0; t < LSTM_SEQ; ++t) {
    // ---- convert current x tile to bf16 A-layout (16x32 per k-step) ----
    // elements 0..7 = K = 32ks+8h+0..7 ; 8..15 = K = 32ks+16+8h+0..7
    v16bf A[2];
    #pragma unroll
    for (int ks = 0; ks < 2; ++ks) {
      v16bf a;
      #pragma unroll
      for (int c = 0; c < 2; ++c) {
        f4 p0 = raw[ks * 4 + c * 2 + 0];
        f4 p1 = raw[ks * 4 + c * 2 + 1];
        #pragma unroll
        for (int e = 0; e < 4; ++e) {
          a[c * 8 + e]     = (__bf16)p0[e];
          a[c * 8 + 4 + e] = (__bf16)p1[e];
        }
      }
      A[ks] = a;
    }

    // ---- prefetch next timestep's x tile (software pipeline) ----
    if (t + 1 < LSTM_SEQ) {
      const float* p = xrow + (size_t)(t + 1) * LSTM_IN;
      #pragma unroll
      for (int ks = 0; ks < 2; ++ks)
        #pragma unroll
        for (int c = 0; c < 2; ++c) {
          const f4* q = (const f4*)(p + ks * 32 + c * 16 + h * 8);
          raw[ks * 4 + c * 2 + 0] = q[0];
          raw[ks * 4 + c * 2 + 1] = q[1];
        }
    }

    // ---- gates: bias + x@Wih^T (bf16 WMMA) + h@Whh^T (f32 WMMA) ----
    v8f acc[4];
    #pragma unroll
    for (int g = 0; g < 4; ++g) {
      v8f cg;
      #pragma unroll
      for (int j = 0; j < 8; ++j) cg[j] = bias[g];
      cg = __builtin_amdgcn_wmma_f32_16x16x32_bf16(false, A[0], false, Bih[g][0],
                                                   (short)0, cg, false, false);
      cg = __builtin_amdgcn_wmma_f32_16x16x32_bf16(false, A[1], false, Bih[g][1],
                                                   (short)0, cg, false, false);
#if __has_builtin(__builtin_amdgcn_wmma_f32_16x16x4_f32)
      #pragma unroll
      for (int ks = 0; ks < 3; ++ks)
        cg = __builtin_amdgcn_wmma_f32_16x16x4_f32(false, Ah[ks], false, Bhh[g][ks],
                                                   (short)0, cg, false, false);
#else
      // fallback: per-lane dot products against hbuf (previous h)
      #pragma unroll
      for (int j = 0; j < 8; ++j) {
        float s = cg[j];
        #pragma unroll
        for (int u = 0; u < LSTM_HID; ++u)
          s = __builtin_fmaf(hbuf[8 * h + j][u], WhhR[g][u], s);
        cg[j] = s;
      }
#endif
      acc[g] = cg;
    }

    // ---- per-lane LSTM cell (gate order i,f,g,o) ----
    float hval[8];
    #pragma unroll
    for (int j = 0; j < 8; ++j) {
      float ig = fast_sigmoid(acc[0][j]);
      float fg = fast_sigmoid(acc[1][j]);
      float gg = fast_tanh   (acc[2][j]);
      float og = fast_sigmoid(acc[3][j]);
      float cn = __builtin_fmaf(fg, cst[j], ig * gg);
      cst[j]  = cn;
      hval[j] = og * fast_tanh(cn);
    }

    // ---- store hs[b][t][:] (and h_last at t==SEQ-1) ----
    if (n < LSTM_HID) {
      #pragma unroll
      for (int j = 0; j < 8; ++j) {
        int m = 8 * h + j;
        out[(((size_t)(b0 + m)) * LSTM_SEQ + t) * LSTM_HID + n] = hval[j];
      }
      if (t == LSTM_SEQ - 1) {
        #pragma unroll
        for (int j = 0; j < 8; ++j) {
          int m = 8 * h + j;
          out[HS_OFF + (size_t)(b0 + m) * LSTM_HID + n] = hval[j];
        }
      }
    }

    // ---- transpose h (C-layout -> A-layout) through LDS ----
    if (n < LSTM_HID) {
      #pragma unroll
      for (int j = 0; j < 8; ++j) hbuf[8 * h + j][n] = hval[j];
    }
    __syncthreads();
#if __has_builtin(__builtin_amdgcn_wmma_f32_16x16x4_f32)
    #pragma unroll
    for (int ks = 0; ks < 3; ++ks) {
      int u0 = 4 * ks + 2 * h;
      Ah[ks][0] = (u0     < LSTM_HID) ? hbuf[r][u0]     : 0.0f;
      Ah[ks][1] = (u0 + 1 < LSTM_HID) ? hbuf[r][u0 + 1] : 0.0f;
    }
#endif
    __syncthreads();
  }
}

extern "C" void kernel_launch(void* const* d_in, const int* in_sizes, int n_in,
                              void* d_out, int out_size, void* d_ws, size_t ws_size,
                              hipStream_t stream) {
  const float* x   = (const float*)d_in[0];
  const float* Wih = (const float*)d_in[1];
  const float* Whh = (const float*)d_in[2];
  const float* bih = (const float*)d_in[3];
  const float* bhh = (const float*)d_in[4];
  float* out = (float*)d_out;
  (void)in_sizes; (void)n_in; (void)out_size; (void)d_ws; (void)ws_size;

  dim3 grid(LSTM_BATCH / 16);   // 256 single-wave workgroups
  dim3 block(32);               // one wave32 per block
  lstm_fused_wmma<<<grid, block, 0, stream>>>(x, Wih, Whh, bih, bhh, out);
}
